// InfoNCELoss_84447646974664
// MI455X (gfx1250) — compile-verified, matrix operand order checked by reading, not run on hardware
//
#include <hip/hip_runtime.h>
#include <hip/hip_bf16.h>
#include <math.h>

typedef __attribute__((ext_vector_type(16))) _Float16 v16h;
typedef __attribute__((ext_vector_type(8)))  _Float16 v8h;
typedef __attribute__((ext_vector_type(8)))  float    v8f;
typedef __attribute__((ext_vector_type(4)))  int      v4i;

#define B_ROWS   4096
#define DIM      512
#define NROWS    (2 * B_ROWS)            // 8192
#define TEMP_INV 10.0f                   // 1 / 0.1
#define NSLICE   8
#define COLS_PER_SLICE (NROWS / NSLICE)  // 1024

#define KCHUNK   256                     // halves of K staged per LDS chunk
#define LDSTRIDE 264                     // 256 + 8 pad halves -> conflict-free ds_load_b128
#define AS1 __attribute__((address_space(1)))
#define AS3 __attribute__((address_space(3)))

#if defined(__AMDGCN__) && __has_builtin(__builtin_amdgcn_global_load_async_to_lds_b128)
#define HAVE_ASYNC_LDS 1
#else
#define HAVE_ASYNC_LDS 0
#endif

__device__ __forceinline__ void async_copy_b128_to_lds(const _Float16* gsrc,
                                                       _Float16* ldst) {
#if HAVE_ASYNC_LDS
  v4i* gtmp = (v4i*)gsrc;                 // strip const, retype (generic AS)
  v4i* ltmp = (v4i*)ldst;
  AS1 v4i* g1 = (AS1 v4i*)gtmp;           // addrspace cast -> global
  AS3 v4i* l3 = (AS3 v4i*)ltmp;           // addrspace cast -> LDS
  __builtin_amdgcn_global_load_async_to_lds_b128(g1, l3, 0, 0);
#else
  *(v8h*)ldst = *(const v8h*)gsrc;
#endif
}

__device__ __forceinline__ void wait_asynccnt0() {
#if defined(__AMDGCN__)
#if __has_builtin(__builtin_amdgcn_s_wait_asynccnt)
  __builtin_amdgcn_s_wait_asynccnt(0);
#else
  asm volatile("s_wait_asynccnt 0x0" ::: "memory");
#endif
#endif
}

// ---------------------------------------------------------------------------
// Kernel 1: per pair-row i, compute norms of f1_i, f2_i and their raw dot in
// one pass; write fp16 normalized rows into F[i] and F[i+B]; positive logit
// pos[i] = pos[i+B] = (f1_i . f2_i) / (|f1_i| |f2_i|)   (x10 applied later).
// ---------------------------------------------------------------------------
__global__ void __launch_bounds__(256)
normalize_pair_kernel(const float* __restrict__ f1,
                      const float* __restrict__ f2,
                      _Float16* __restrict__ F,
                      float* __restrict__ pos) {
  const int i = blockIdx.x;        // 0..B_ROWS-1
  const int t = threadIdx.x;       // 0..255
  const float* a = f1 + (size_t)i * DIM;
  const float* b = f2 + (size_t)i * DIM;
  float a0 = a[t], a1 = a[t + 256];
  float b0 = b[t], b1 = b[t + 256];

  float s1 = a0 * a0 + a1 * a1;
  float s2 = b0 * b0 + b1 * b1;
  float sd = a0 * b0 + a1 * b1;

  #pragma unroll
  for (int off = 16; off; off >>= 1) {
    s1 += __shfl_xor(s1, off, 32);
    s2 += __shfl_xor(s2, off, 32);
    sd += __shfl_xor(sd, off, 32);
  }
  __shared__ float r1[8], r2[8], rd[8];
  __shared__ float inv1s, inv2s;
  const int lane = t & 31, wv = t >> 5;
  if (lane == 0) { r1[wv] = s1; r2[wv] = s2; rd[wv] = sd; }
  __syncthreads();
  if (t == 0) {
    float S1 = 0.f, S2 = 0.f, SD = 0.f;
    #pragma unroll
    for (int w = 0; w < 8; ++w) { S1 += r1[w]; S2 += r2[w]; SD += rd[w]; }
    float n1 = fmaxf(sqrtf(S1), 1e-12f);
    float n2 = fmaxf(sqrtf(S2), 1e-12f);
    float i1 = 1.0f / n1, i2 = 1.0f / n2;
    inv1s = i1; inv2s = i2;
    float d = SD * i1 * i2;
    pos[i] = d;
    pos[i + B_ROWS] = d;
  }
  __syncthreads();
  const float i1 = inv1s, i2 = inv2s;
  _Float16* Fa = F + (size_t)i * DIM;
  _Float16* Fb = F + (size_t)(i + B_ROWS) * DIM;
  Fa[t]       = (_Float16)(a0 * i1);
  Fa[t + 256] = (_Float16)(a1 * i1);
  Fb[t]       = (_Float16)(b0 * i2);
  Fb[t + 256] = (_Float16)(b1 * i2);
}

// ---------------------------------------------------------------------------
// Kernel 2: fused sim = (F F^T)/T with streaming exp row-sum, diag masked.
//
// Block = 8 waves, all on the same column slice. Per 64-column pass the block
// stages the B panel (64 rows of F) in LDS in two 256-half K-chunks using
// async global->LDS copies (ASYNCcnt), then every wave computes a 32x64 tile:
// per K-step 2 A fragments (global, L0-resident) x 4 shared B fragments
// (ds_load_b128 from padded LDS) -> 8 v_wmma_f32_16x16x32_f16.
// Epilogue applies 1/T, masks the diagonal, accumulates exp per row.
//
// Fragment layouts (CDNA5 ISA 7.12.2, 16-bit A 16x32): lane L<16 holds
// M=L, K in {ks..ks+7, ks+16..ks+23}; lane L>=16 holds M=L-16, K offset +8.
// B of F*F^T has the same per-lane pattern with the column tile as row base.
// ---------------------------------------------------------------------------
__global__ void __launch_bounds__(256)
simexp_kernel(const _Float16* __restrict__ F, float* __restrict__ partial) {
  __shared__ _Float16 ldsB[64 * LDSTRIDE];   // ~33 KB

  const int t      = threadIdx.x;
  const int lane   = t & 31;
  const int wv     = t >> 5;            // wave in block: 0..7
  const int laneLo = lane & 15;
  const int laneHi = lane >> 4;         // 0 or 1

  const int blk      = blockIdx.x;      // 0..255
  const int slice    = blk & 7;
  const int rowBlock = blk >> 3;        // 0..31
  const int rowBase  = (rowBlock * 8 + wv) * 32;  // this wave's 32-row base

  const _Float16* a0row = F + (size_t)(rowBase + laneLo) * DIM + laneHi * 8;
  const _Float16* a1row = a0row + (size_t)16 * DIM;

  float eacc[16];
  #pragma unroll
  for (int v = 0; v < 16; ++v) eacc[v] = 0.f;

  const int colStart = slice * COLS_PER_SLICE;
  const int colEnd   = colStart + COLS_PER_SLICE;

  for (int colBase = colStart; colBase < colEnd; colBase += 64) {
    v8f c[2][4] = {};

    for (int kc = 0; kc < DIM; kc += KCHUNK) {
      // ---- cooperative staging: 64 cols x 256 halves (2048 x 16B segments)
      #pragma unroll
      for (int j = 0; j < 8; ++j) {
        const int s_lin = j * 256 + t;
        const int cc    = s_lin >> 5;        // staged column 0..63
        const int seg   = s_lin & 31;        // 16B segment within chunk row
        const _Float16* gsrc =
            F + (size_t)(colBase + cc) * DIM + kc + seg * 8;
        _Float16* ldst = &ldsB[cc * LDSTRIDE + seg * 8];
        async_copy_b128_to_lds(gsrc, ldst);
      }
#if HAVE_ASYNC_LDS
      wait_asynccnt0();
#endif
      __syncthreads();

      // ---- 8 K-steps over the staged chunk
      #pragma unroll 2
      for (int ks = 0; ks < KCHUNK; ks += 32) {
        v16h a0, a1;
        {
          const _Float16* pa0 = a0row + kc + ks;
          const _Float16* pa1 = a1row + kc + ks;
          *((v8h*)&a0)     = *(const v8h*)(pa0);
          *((v8h*)&a0 + 1) = *(const v8h*)(pa0 + 16);
          *((v8h*)&a1)     = *(const v8h*)(pa1);
          *((v8h*)&a1 + 1) = *(const v8h*)(pa1 + 16);
        }
        #pragma unroll
        for (int tl = 0; tl < 4; ++tl) {
          const _Float16* bl =
              &ldsB[(tl * 16 + laneLo) * LDSTRIDE + ks + laneHi * 8];
          v16h bf;
          *((v8h*)&bf)     = *(const v8h*)(bl);
          *((v8h*)&bf + 1) = *(const v8h*)(bl + 16);
          c[0][tl] = __builtin_amdgcn_wmma_f32_16x16x32_f16(
              false, a0, false, bf, (short)0, c[0][tl], false, false);
          c[1][tl] = __builtin_amdgcn_wmma_f32_16x16x32_f16(
              false, a1, false, bf, (short)0, c[1][tl], false, false);
        }
      }
      __syncthreads();   // LDS reusable for next chunk / pass
    }

    // ---- epilogue: scale by 1/T, mask diagonal, accumulate exp per row
    #pragma unroll
    for (int mt = 0; mt < 2; ++mt) {
      #pragma unroll
      for (int tl = 0; tl < 4; ++tl) {
        const int gc = colBase + tl * 16 + laneLo;
        #pragma unroll
        for (int v = 0; v < 8; ++v) {
          const int gr = rowBase + mt * 16 + v + laneHi * 8;
          const float s = c[mt][tl][v] * TEMP_INV;
          if (gr != gc) eacc[mt * 8 + v] += __expf(s);
        }
      }
    }
  }

  // Reduce across the 16 lanes of each N-group (xor within bits 0..3 only).
  #pragma unroll
  for (int v = 0; v < 16; ++v) {
    float x = eacc[v];
    x += __shfl_xor(x, 1, 32);
    x += __shfl_xor(x, 2, 32);
    x += __shfl_xor(x, 4, 32);
    x += __shfl_xor(x, 8, 32);
    eacc[v] = x;
  }

  // lane 0 holds rows {mt*16 + v}, lane 16 holds rows {mt*16 + 8 + v}.
  if (laneLo == 0) {
    #pragma unroll
    for (int mt = 0; mt < 2; ++mt) {
      float* part = partial + (size_t)slice * NROWS + rowBase + mt * 16 + laneHi * 8;
      #pragma unroll
      for (int v = 0; v < 8; ++v) part[v] = eacc[mt * 8 + v];
    }
  }
}

// ---------------------------------------------------------------------------
// Kernel 3: deterministic final reduction.
// loss = mean_i ( log(S_i) - pos_i * 10 ),  S_i = sum over 8 slices.
// ---------------------------------------------------------------------------
__global__ void __launch_bounds__(256)
finalize_kernel(const float* __restrict__ partial,
                const float* __restrict__ pos,
                float* __restrict__ out) {
  const int t = threadIdx.x;   // single block of 256
  float acc = 0.f;
  for (int r = t; r < NROWS; r += 256) {
    float S = 0.f;
    #pragma unroll
    for (int s = 0; s < NSLICE; ++s) S += partial[(size_t)s * NROWS + r];
    acc += logf(S) - pos[r] * TEMP_INV;
  }
  #pragma unroll
  for (int off = 16; off; off >>= 1) acc += __shfl_xor(acc, off, 32);
  __shared__ float red[8];
  const int lane = t & 31, wv = t >> 5;
  if (lane == 0) red[wv] = acc;
  __syncthreads();
  if (t == 0) {
    float tot = 0.f;
    #pragma unroll
    for (int w = 0; w < 8; ++w) tot += red[w];
    out[0] = tot / (float)NROWS;
  }
}

// ---------------------------------------------------------------------------
extern "C" void kernel_launch(void* const* d_in, const int* in_sizes, int n_in,
                              void* d_out, int out_size, void* d_ws, size_t ws_size,
                              hipStream_t stream) {
  (void)in_sizes; (void)n_in; (void)out_size; (void)ws_size;
  const float* f1 = (const float*)d_in[0];
  const float* f2 = (const float*)d_in[1];
  float* out = (float*)d_out;

  char* ws = (char*)d_ws;
  _Float16* F   = (_Float16*)ws;                                   // 8 MB
  float*    pos = (float*)(ws + (size_t)NROWS * DIM * sizeof(_Float16));
  float*    partial = (float*)(ws + (size_t)NROWS * DIM * sizeof(_Float16)
                                  + (size_t)NROWS * sizeof(float)); // 256 KB

  normalize_pair_kernel<<<B_ROWS, 256, 0, stream>>>(f1, f2, F, pos);

  // 256 blocks x 8 waves; each wave: 32 rows x one 1024-col slice.
  simexp_kernel<<<256, 256, 0, stream>>>(F, partial);

  finalize_kernel<<<1, 256, 0, stream>>>(partial, pos, out);
}
